// Head_19971597926765
// MI455X (gfx1250) — matrix operand order repaired and checked by reading, not compiled, not run on hardware
//
#include <hip/hip_runtime.h>
#include <hip/hip_bf16.h>
#include <cstdint>

// ---------------- problem constants ----------------
#define B_  128
#define L_  49
#define C_  512
#define N_  27
#define NS_ 14
#define M1  (B_*L_)   // 6272 rows of the big GEMM
#define M2  (B_*N_)   // 3456 rows of the VU GEMM
#define NC  (N_*C_)   // 13824 BN channels

typedef __attribute__((ext_vector_type(16))) __bf16 v16bf;
typedef __attribute__((ext_vector_type(8)))  float  v8f;
typedef __attribute__((ext_vector_type(4)))  uint32_t v4u;
typedef __attribute__((ext_vector_type(8)))  int  v8i;
typedef __attribute__((ext_vector_type(4)))  int  v4i;

// ---------------- TDM availability / arity detection ----------------
#if defined(__has_builtin)
# if __has_builtin(__builtin_amdgcn_tensor_load_to_lds)
#  define HAVE_TDM 1
# else
#  define HAVE_TDM 0
# endif
#else
# define HAVE_TDM 0
#endif
#if defined(__has_include)
# if __has_include(<hip/amd_detail/amd_gfx1250_TDM.h>)
#  define TDM_SIX_ARGS 1      // therock-10.0 headers => clang-23 6-arg builtin
# else
#  define TDM_SIX_ARGS 0     // ROCm 7.2 clang-22 5-arg builtin
# endif
#else
# define TDM_SIX_ARGS 0
#endif

// ---------------- workspace layout (bytes) ----------------
static constexpr size_t AL(size_t x){ return (x + 255) & ~size_t(255); }
constexpr size_t OFF_XB   = 0;                                   // bf16 [M1][C]
constexpr size_t OFF_WB   = AL(OFF_XB   + (size_t)M1*C_*2);      // bf16 [N][C][C]
constexpr size_t OFF_WCAT = AL(OFF_WB   + (size_t)N_*C_*C_*2);   // bf16 [1024][C]  (V rows, then U rows)
constexpr size_t OFF_BCAT = AL(OFF_WCAT + (size_t)1024*C_*2);    // f32  [1024]
constexpr size_t OFF_GSUM = AL(OFF_BCAT + 1024*4);               // f32  [NC]
constexpr size_t OFF_GSQ  = OFF_GSUM + (size_t)NC*4;             // f32  [NC]  (contiguous w/ GSUM for one memset)
constexpr size_t OFF_SC   = AL(OFF_GSQ  + (size_t)NC*4);         // f32  [NC]  BN scale
constexpr size_t OFF_SH   = AL(OFF_SC   + (size_t)NC*4);         // f32  [NC]  BN shift
constexpr size_t OFF_FV   = AL(OFF_SH   + (size_t)NC*4);         // f32  [B][N][C]
constexpr size_t OFF_FVB  = AL(OFF_FV   + (size_t)M2*C_*4);      // bf16 [B][N][C]
constexpr size_t OFF_A    = AL(OFF_FVB  + (size_t)M2*C_*2);      // f32  [B][N][N]
constexpr size_t OFF_VU   = AL(OFF_A    + (size_t)B_*N_*N_*4);   // f32  [M2][1024]
constexpr size_t OFF_Z    = AL(OFF_VU   + (size_t)M2*1024*4);    // f32  [M2][C]
constexpr size_t OFF_BNV  = AL(OFF_Z    + (size_t)M2*C_*4);      // f32  [N][2] scale,shift
constexpr size_t OFF_MSC  = AL(OFF_BNV  + N_*2*4);               // f32  [27][C]
constexpr size_t OFF_SSC  = AL(OFF_MSC  + (size_t)N_*C_*4);      // f32  [14][C]
constexpr size_t OFF_FPRE = AL(OFF_SSC  + (size_t)NS_*C_*4);     // bf16 [N][M1][C]

// ---------------- WMMA fragment loaders (CDNA5 16-bit layouts) ----------------
// LDS tiles are [128 rows][32 K] padded to 40 halves/row (80B: 16B-aligned fills, 8B-aligned frag reads).
#define TPITCH 40

// A 16x32: lanes 0-15 -> M=lane, K in {0..7,16..23}; lanes 16-31 -> K in {8..15,24..31}
__device__ __forceinline__ v16bf frag_a(const __bf16* t, int row, int lane){
  const int kb = (lane & 16) ? 8 : 0;
  const __bf16* p = t + row*TPITCH + kb;
  union { v16bf v; uint64_t q[4]; } f;
  f.q[0] = *(const uint64_t*)(p);
  f.q[1] = *(const uint64_t*)(p + 4);
  f.q[2] = *(const uint64_t*)(p + 16);
  f.q[3] = *(const uint64_t*)(p + 20);
  return f.v;
}
// B 32x16: lanes 0-15 -> N=lane, K in {0..7,16..23}; lanes 16-31 -> K in {8..15,24..31}
__device__ __forceinline__ v16bf frag_b(const __bf16* t, int row, int lane){
  const int kb = (lane & 16) ? 8 : 0;
  const __bf16* p = t + row*TPITCH + kb;
  union { v16bf v; uint64_t q[4]; } f;
  f.q[0] = *(const uint64_t*)(p);
  f.q[1] = *(const uint64_t*)(p + 4);
  f.q[2] = *(const uint64_t*)(p + 16);
  f.q[3] = *(const uint64_t*)(p + 20);
  return f.v;
}

__device__ __forceinline__ float blockReduce(float v, float* buf){
  const int t = threadIdx.x;
  buf[t] = v; __syncthreads();
  for (int s = 128; s > 0; s >>= 1){ if (t < s) buf[t] += buf[t + s]; __syncthreads(); }
  float r = buf[0]; __syncthreads();
  return r;
}

// ---------------- TDM helpers ----------------
__device__ __forceinline__ void wait_tdm(){
#if defined(__has_builtin) && __has_builtin(__builtin_amdgcn_s_wait_tensorcnt)
  __builtin_amdgcn_s_wait_tensorcnt(0);
#else
  asm volatile("s_wait_tensorcnt 0x0" ::: "memory");
#endif
}

#if HAVE_TDM
// DMA a 128-row x 32-col bf16 tile (row stride = C_ elements) from global into LDS,
// with TDM LDS padding of 4 DWORDs every 16 DWORDs => 40-half row pitch (TPITCH).
__device__ __forceinline__ void tdm_load_tile(const __bf16* gsrc, __bf16* ldst){
  const uint32_t lds = (uint32_t)(uintptr_t)ldst;         // flat->LDS: low 32 bits
  const uint64_t ga  = (uint64_t)(uintptr_t)gsrc;
  // group0: count=1 | lds_addr | global_addr[56:0] | type=2
  v4u g0 = { 1u, lds, (uint32_t)ga,
             (uint32_t)((ga >> 32) & 0x1FFFFFFu) | (2u << 30) };
  // group1: data_size=2B, pad_enable, pad_interval=3 (16 DW), pad_amount=3 (4 DW)
  //         tensor_dim0=512, tensor_dim1=128, tile_dim0=32, tile_dim1=128,
  //         tensor_dim0_stride=512
  v8i g1 = { (int)((1u << 16) | (1u << 20) | (3u << 22) | (3u << 25)),
             (int)((uint32_t)C_ << 16),   // tensor_dim0 low16 at bits[63:48]
             (int)(128u << 16),           // tensor_dim0 hi=0 | tensor_dim1 low16
             (int)(32u  << 16),           // tensor_dim1 hi=0 | tile_dim0=32
             (int)128,                    // tile_dim1=128 | tile_dim2=0
             (int)C_,                     // tensor_dim0_stride low32
             0, 0 };
  v4i z4 = {};
#if TDM_SIX_ARGS
  v8i z8 = {};
  __builtin_amdgcn_tensor_load_to_lds(g0, g1, z4, z4, z8, 0);
#else
  __builtin_amdgcn_tensor_load_to_lds(g0, g1, z4, z4, 0);
#endif
}
#endif

// ---------------- K1: precision conversion / weight packing ----------------
__global__ __launch_bounds__(256) void k_convert(const float* x, const float* linW,
    const float* Vw, const float* Vb, const float* Uw, const float* Ub,
    __bf16* xb, __bf16* wb, __bf16* wcat, float* bcat)
{
  size_t i = (size_t)blockIdx.x * 256 + threadIdx.x;
  if (i < (size_t)M1*C_)      xb[i] = (__bf16)x[i];
  if (i < (size_t)N_*C_*C_)   wb[i] = (__bf16)linW[i];
  if (i < (size_t)2*C_*C_)    wcat[i] = (__bf16)((i < (size_t)C_*C_) ? Vw[i] : Uw[i - (size_t)C_*C_]);
  if (i < 1024)               bcat[i] = (i < 512) ? Vb[i] : Ub[i - 512];
}

// ---------------- shared GEMM inner step: 8 WMMAs per wave on one LDS buffer ----------------
__device__ __forceinline__ void wmma_step(const __bf16* At, const __bf16* Bt,
                                          int wm, int wn, int lane, v8f acc[4][2])
{
  v16bf bf[2];
  #pragma unroll
  for (int ns = 0; ns < 2; ++ns)
    bf[ns] = frag_b(Bt, wn*32 + ns*16 + (lane & 15), lane);
  #pragma unroll
  for (int ms = 0; ms < 4; ++ms){
    v16bf af = frag_a(At, wm*64 + ms*16 + (lane & 15), lane);
    #pragma unroll
    for (int ns = 0; ns < 2; ++ns)
      acc[ms][ns] = __builtin_amdgcn_wmma_f32_16x16x32_bf16(
          false, af, false, bf[ns], (short)0, acc[ms][ns], false, false);
  }
}

// ---------------- K2: main WMMA GEMM  f[n][m][d] = x @ W_n^T + b_n, fused BN stats ----------------
__global__ __launch_bounds__(256) void k_gemm_f(const __bf16* __restrict__ xb,
    const __bf16* __restrict__ wb, const float* __restrict__ lin_b,
    __bf16* __restrict__ fpre, float* __restrict__ gsum, float* __restrict__ gsq)
{
  __shared__ __bf16 At[2][128*TPITCH];
  __shared__ __bf16 Bt[2][128*TPITCH];
  __shared__ float  csum[128], csq[128];

  const int n  = blockIdx.z;
  const int m0 = blockIdx.x * 128;
  const int d0 = blockIdx.y * 128;
  const int tid = threadIdx.x, lane = tid & 31, wid = tid >> 5;
  const int wm = wid >> 2, wn = wid & 3;          // 2x4 wave grid, each wave: 64(M) x 32(N)
  const __bf16* wsrc = wb + (size_t)n * C_ * C_;
  const __bf16* Asrc = xb   + (size_t)m0 * C_;
  const __bf16* Bsrc = wsrc + (size_t)d0 * C_;

  v8f acc[4][2];
  #pragma unroll
  for (int ms = 0; ms < 4; ++ms)
    #pragma unroll
    for (int ns = 0; ns < 2; ++ns){
      float bv = lin_b[n*C_ + d0 + wn*32 + ns*16 + (lane & 15)];
      #pragma unroll
      for (int v = 0; v < 8; ++v) acc[ms][ns][v] = bv;
    }

#if HAVE_TDM
  // TDM double-buffered pipeline: wave0 DMAs tile k+1 while all waves run WMMAs on tile k.
  if (wid == 0){
    tdm_load_tile(Asrc, &At[0][0]);
    tdm_load_tile(Bsrc, &Bt[0][0]);
    wait_tdm();
  }
  __syncthreads();
  for (int kt = 0; kt < C_/32; ++kt){
    const int cur = kt & 1, nxt = cur ^ 1;
    if (wid == 0 && kt + 1 < C_/32){
      tdm_load_tile(Asrc + (kt+1)*32, &At[nxt][0]);
      tdm_load_tile(Bsrc + (kt+1)*32, &Bt[nxt][0]);
    }
    wmma_step(&At[cur][0], &Bt[cur][0], wm, wn, lane, acc);
    if (wid == 0) wait_tdm();
    __syncthreads();
  }
#else
  for (int kt = 0; kt < C_/32; ++kt){
    const int k0 = kt * 32;
    for (int c = tid; c < 512; c += 256){
      int row = c >> 2, cc = c & 3;
      *(uint4*)(&At[0][0] + row*TPITCH + cc*8) = *(const uint4*)(Asrc + (size_t)row*C_ + k0 + cc*8);
      *(uint4*)(&Bt[0][0] + row*TPITCH + cc*8) = *(const uint4*)(Bsrc + (size_t)row*C_ + k0 + cc*8);
    }
    __syncthreads();
    if (kt + 1 < C_/32){
      __builtin_prefetch(Asrc + (size_t)(tid >> 1)*C_ + k0 + 32, 0, 1);
      __builtin_prefetch(Bsrc + (size_t)(tid >> 1)*C_ + k0 + 32, 0, 1);
    }
    wmma_step(&At[0][0], &Bt[0][0], wm, wn, lane, acc);
    __syncthreads();
  }
#endif

  // epilogue: store f (bf16) + per-column sum/sumsq for BatchNorm (stats over b,l)
  if (tid < 128){ csum[tid] = 0.f; csq[tid] = 0.f; }
  __syncthreads();
  const int rbase = m0 + wm*64 + ((lane & 16) ? 8 : 0);
  #pragma unroll
  for (int ms = 0; ms < 4; ++ms)
    #pragma unroll
    for (int ns = 0; ns < 2; ++ns){
      int col_l = wn*32 + ns*16 + (lane & 15);
      size_t base = ((size_t)n*M1 + rbase + ms*16)*C_ + d0 + col_l;
      float s = 0.f, ss = 0.f;
      #pragma unroll
      for (int v = 0; v < 8; ++v){
        float xv = acc[ms][ns][v];
        fpre[base + (size_t)v*C_] = (__bf16)xv;
        s += xv; ss += xv*xv;
      }
      atomicAdd(&csum[col_l], s);   // ds_add_f32
      atomicAdd(&csq[col_l], ss);
    }
  __syncthreads();
  if (tid < 128){
    atomicAdd(&gsum[n*C_ + d0 + tid], csum[tid]);  // global_atomic_add_f32
    atomicAdd(&gsq [n*C_ + d0 + tid], csq [tid]);
  }
}

// ---------------- K3: finalize BN affine per (n,d) ----------------
__global__ __launch_bounds__(256) void k_bnstats(const float* gsum, const float* gsq,
    const float* bn_w, const float* bn_b, float* scale, float* shift)
{
  int i = blockIdx.x*256 + threadIdx.x;
  if (i < NC){
    float mu  = gsum[i] / (float)M1;
    float var = gsq[i]  / (float)M1 - mu*mu;
    float sc  = bn_w[i] * rsqrtf(var + 1e-5f);
    scale[i] = sc;
    shift[i] = bn_b[i] - mu*sc;
  }
}

// ---------------- K4: BN + relu + mean over L -> f_v ----------------
__global__ __launch_bounds__(256) void k_bnapply(const __bf16* fpre, const float* scale,
    const float* shift, float* fv, __bf16* fvb)
{
  int b = blockIdx.x / N_, n = blockIdx.x % N_;
  for (int c = threadIdx.x; c < C_; c += 256){
    float sc = scale[n*C_ + c], sh = shift[n*C_ + c];
    const __bf16* p = fpre + ((size_t)n*M1 + (size_t)b*L_)*C_ + c;
    float a = 0.f;
    for (int l = 0; l < L_; ++l){
      float f = (float)p[(size_t)l*C_];
      a += fmaxf(f*sc + sh, 0.f);
    }
    a *= (1.f / (float)L_);
    size_t o = ((size_t)b*N_ + n)*C_ + c;
    fv[o]  = a;
    fvb[o] = (__bf16)a;
  }
}

// ---------------- K5: per-batch dynamic top-K graph -> normalized adjacency A ----------------
__global__ __launch_bounds__(256) void k_graph(const float* fv, float* Ag)
{
  __shared__ float fvL[N_*C_];     // 55.3 KB
  __shared__ float si[N_*N_];
  __shared__ float thr[N_];
  __shared__ float dinv[N_];
  const int b = blockIdx.x, tid = threadIdx.x;
  for (int i = tid; i < N_*C_; i += 256) fvL[i] = fv[(size_t)b*N_*C_ + i];
  __syncthreads();
  for (int e = tid; e < N_*N_; e += 256){
    int i = e / N_, j = e % N_;
    float s = 0.f;
    for (int c = 0; c < C_; ++c) s += fvL[i*C_ + c] * fvL[j*C_ + c];
    si[e] = s;
  }
  __syncthreads();
  if (tid < N_){
    // kth(=4th) largest incl. duplicates: max{ v : #(row >= v) >= K }
    float best = -1e30f;
    for (int j = 0; j < N_; ++j){
      float v = si[tid*N_ + j];
      int cnt = 0;
      for (int j2 = 0; j2 < N_; ++j2) cnt += (si[tid*N_ + j2] >= v);
      if (cnt >= 4 && v > best) best = v;
    }
    thr[tid] = best;
    float deg = 0.f;
    for (int j = 0; j < N_; ++j) deg += (si[tid*N_ + j] >= best) ? 1.f : 0.f;
    dinv[tid] = rsqrtf(deg);
  }
  __syncthreads();
  for (int e = tid; e < N_*N_; e += 256){
    int i = e / N_, j = e % N_;
    Ag[(size_t)b*N_*N_ + e] = (si[e] >= thr[i]) ? dinv[i]*dinv[j] : 0.f;
  }
}

// ---------------- K6: WMMA GEMM  VU[m2][0:512]=Vx, [512:1024]=Ux ----------------
__global__ __launch_bounds__(256) void k_gemm_vu(const __bf16* __restrict__ fvb,
    const __bf16* __restrict__ wcat, const float* __restrict__ bcat, float* __restrict__ vu)
{
  __shared__ __bf16 At[2][128*TPITCH];
  __shared__ __bf16 Bt[2][128*TPITCH];
  const int m0 = blockIdx.x * 128;
  const int d0 = blockIdx.y * 128;
  const int tid = threadIdx.x, lane = tid & 31, wid = tid >> 5;
  const int wm = wid >> 2, wn = wid & 3;
  const __bf16* Asrc = fvb  + (size_t)m0 * C_;
  const __bf16* Bsrc = wcat + (size_t)d0 * C_;

  v8f acc[4][2];
  #pragma unroll
  for (int ms = 0; ms < 4; ++ms)
    #pragma unroll
    for (int ns = 0; ns < 2; ++ns){
      float bv = bcat[d0 + wn*32 + ns*16 + (lane & 15)];
      #pragma unroll
      for (int v = 0; v < 8; ++v) acc[ms][ns][v] = bv;
    }

#if HAVE_TDM
  if (wid == 0){
    tdm_load_tile(Asrc, &At[0][0]);
    tdm_load_tile(Bsrc, &Bt[0][0]);
    wait_tdm();
  }
  __syncthreads();
  for (int kt = 0; kt < C_/32; ++kt){
    const int cur = kt & 1, nxt = cur ^ 1;
    if (wid == 0 && kt + 1 < C_/32){
      tdm_load_tile(Asrc + (kt+1)*32, &At[nxt][0]);
      tdm_load_tile(Bsrc + (kt+1)*32, &Bt[nxt][0]);
    }
    wmma_step(&At[cur][0], &Bt[cur][0], wm, wn, lane, acc);
    if (wid == 0) wait_tdm();
    __syncthreads();
  }
#else
  for (int kt = 0; kt < C_/32; ++kt){
    const int k0 = kt * 32;
    for (int c = tid; c < 512; c += 256){
      int row = c >> 2, cc = c & 3;
      *(uint4*)(&At[0][0] + row*TPITCH + cc*8) = *(const uint4*)(Asrc + (size_t)row*C_ + k0 + cc*8);
      *(uint4*)(&Bt[0][0] + row*TPITCH + cc*8) = *(const uint4*)(Bsrc + (size_t)row*C_ + k0 + cc*8);
    }
    __syncthreads();
    wmma_step(&At[0][0], &Bt[0][0], wm, wn, lane, acc);
    __syncthreads();
  }
#endif

  const int rbase = m0 + wm*64 + ((lane & 16) ? 8 : 0);
  #pragma unroll
  for (int ms = 0; ms < 4; ++ms)
    #pragma unroll
    for (int ns = 0; ns < 2; ++ns){
      int col = d0 + wn*32 + ns*16 + (lane & 15);
      size_t base = (size_t)(rbase + ms*16)*1024 + col;
      #pragma unroll
      for (int v = 0; v < 8; ++v) vu[base + (size_t)v*1024] = acc[ms][ns][v];
    }
}

// ---------------- K7: z = A @ Vx + Ux ----------------
__global__ __launch_bounds__(256) void k_z(const float* vu, const float* Ag, float* zg)
{
  __shared__ float VxL[N_*C_];
  __shared__ float AL_[N_*N_];
  const int b = blockIdx.x, tid = threadIdx.x;
  for (int i = tid; i < N_*C_; i += 256){
    int j = i / C_, c = i & (C_-1);
    VxL[i] = vu[((size_t)(b*N_ + j))*1024 + c];
  }
  for (int e = tid; e < N_*N_; e += 256) AL_[e] = Ag[(size_t)b*N_*N_ + e];
  __syncthreads();
  for (int c = tid; c < C_; c += 256){
    for (int i = 0; i < N_; ++i){
      float a = vu[((size_t)(b*N_ + i))*1024 + 512 + c];  // Ux
      for (int j = 0; j < N_; ++j) a += AL_[i*N_ + j] * VxL[j*C_ + c];
      zg[((size_t)(b*N_ + i))*C_ + c] = a;
    }
  }
}

// ---------------- K8: BN over (b,c) per n for z ----------------
__global__ __launch_bounds__(256) void k_bnv(const float* zg, const float* w,
    const float* bvec, float* bnv)
{
  __shared__ float rs[256], rq[256];
  const int n = blockIdx.x, t = threadIdx.x;
  float s = 0.f, sq = 0.f;
  for (int i = t; i < B_*C_; i += 256){
    int b = i >> 9, c = i & 511;
    float v = zg[((size_t)(b*N_ + n))*C_ + c];
    s += v; sq += v*v;
  }
  rs[t] = s; rq[t] = sq; __syncthreads();
  for (int o = 128; o > 0; o >>= 1){ if (t < o){ rs[t] += rs[t+o]; rq[t] += rq[t+o]; } __syncthreads(); }
  if (t == 0){
    const float cnt = (float)(B_*C_);
    float mu = rs[0]/cnt, var = rq[0]/cnt - mu*mu;
    float scl = w[n] * rsqrtf(var + 1e-5f);
    bnv[2*n] = scl; bnv[2*n+1] = bvec[n] - mu*scl;
  }
}

// ---------------- K9: normalize class centers ----------------
__global__ __launch_bounds__(256) void k_normsc(const float* mainsc, const float* subsc,
    float* msc, float* ssc)
{
  __shared__ float red[256];
  const int r = blockIdx.x, t = threadIdx.x;
  const float* src = (r < N_) ? (mainsc + (size_t)r*C_) : (subsc + (size_t)(r - N_)*C_);
  float* dst       = (r < N_) ? (msc    + (size_t)r*C_) : (ssc    + (size_t)(r - N_)*C_);
  float v0 = fmaxf(src[t], 0.f), v1 = fmaxf(src[t + 256], 0.f);
  float nsq = blockReduce(v0*v0 + v1*v1, red);
  float rn = 1.f / fmaxf(sqrtf(nsq), 1e-12f);
  dst[t] = v0*rn; dst[t + 256] = v1*rn;
}

// ---------------- K10: residual + l2norm + logits ----------------
__global__ __launch_bounds__(256) void k_final(const float* zg, const float* fv,
    const float* bnv, const float* msc, const float* ssc, float* out)
{
  __shared__ float red[256];
  const int b = blockIdx.x / N_, n = blockIdx.x % N_;
  const int t = threadIdx.x;
  const float sc = bnv[2*n], sh = bnv[2*n+1];
  const size_t base = ((size_t)(b*N_ + n))*C_;
  float f0 = fmaxf(fv[base + t]       + (zg[base + t]      *sc + sh), 0.f);
  float f1 = fmaxf(fv[base + t + 256] + (zg[base + t + 256]*sc + sh), 0.f);
  float nsq = blockReduce(f0*f0 + f1*f1, red);
  float rn = 1.f / fmaxf(sqrtf(nsq), 1e-12f);
  float dm = blockReduce(f0*msc[n*C_ + t] + f1*msc[n*C_ + t + 256], red) * rn;
  if (t == 0) out[b*41 + n] = dm;
  // SUB_LIST = {0,1,2,4,7,8,11}
  const int subIdx[N_] = {0,1,2,-1,3,-1,-1,4,5,-1,-1,6,-1,-1,-1,-1,-1,-1,-1,-1,-1,-1,-1,-1,-1,-1,-1};
  const int si = subIdx[n];               // uniform across block -> barriers safe
  if (si >= 0){
    float d0v = blockReduce(f0*ssc[(2*si  )*C_ + t] + f1*ssc[(2*si  )*C_ + t + 256], red) * rn;
    float d1v = blockReduce(f0*ssc[(2*si+1)*C_ + t] + f1*ssc[(2*si+1)*C_ + t + 256], red) * rn;
    if (t == 0){ out[b*41 + 27 + 2*si] = d0v; out[b*41 + 27 + 2*si + 1] = d1v; }
  }
}

// ---------------- launcher ----------------
extern "C" void kernel_launch(void* const* d_in, const int* in_sizes, int n_in,
                              void* d_out, int out_size, void* d_ws, size_t ws_size,
                              hipStream_t stream)
{
  (void)in_sizes; (void)n_in; (void)out_size; (void)ws_size;
  const float* x       = (const float*)d_in[0];
  const float* lin_W   = (const float*)d_in[1];
  const float* lin_b   = (const float*)d_in[2];
  const float* lbn_w   = (const float*)d_in[3];
  const float* lbn_b   = (const float*)d_in[4];
  const float* U_w     = (const float*)d_in[5];
  const float* U_b     = (const float*)d_in[6];
  const float* V_w     = (const float*)d_in[7];
  const float* V_b     = (const float*)d_in[8];
  const float* bnv_w   = (const float*)d_in[9];
  const float* bnv_b   = (const float*)d_in[10];
  const float* main_sc = (const float*)d_in[11];
  const float* sub_sc  = (const float*)d_in[12];
  float* out = (float*)d_out;

  char* ws = (char*)d_ws;
  __bf16* xb   = (__bf16*)(ws + OFF_XB);
  __bf16* wb   = (__bf16*)(ws + OFF_WB);
  __bf16* wcat = (__bf16*)(ws + OFF_WCAT);
  float*  bcat = (float*)(ws + OFF_BCAT);
  float*  gsum = (float*)(ws + OFF_GSUM);
  float*  gsq  = (float*)(ws + OFF_GSQ);
  float*  scl  = (float*)(ws + OFF_SC);
  float*  shf  = (float*)(ws + OFF_SH);
  float*  fv   = (float*)(ws + OFF_FV);
  __bf16* fvb  = (__bf16*)(ws + OFF_FVB);
  float*  Ag   = (float*)(ws + OFF_A);
  float*  vu   = (float*)(ws + OFF_VU);
  float*  zg   = (float*)(ws + OFF_Z);
  float*  bnv  = (float*)(ws + OFF_BNV);
  float*  msc  = (float*)(ws + OFF_MSC);
  float*  ssc  = (float*)(ws + OFF_SSC);
  __bf16* fpre = (__bf16*)(ws + OFF_FPRE);

  hipMemsetAsync(ws + OFF_GSUM, 0, (size_t)NC*4*2, stream);  // gsum+gsq contiguous

  const unsigned cvtBlocks = (unsigned)(((size_t)N_*C_*C_ + 255) / 256);  // largest converted array
  k_convert<<<cvtBlocks, 256, 0, stream>>>(x, lin_W, V_w, V_b, U_w, U_b, xb, wb, wcat, bcat);
  k_gemm_f<<<dim3(M1/128, C_/128, N_), 256, 0, stream>>>(xb, wb, lin_b, fpre, gsum, gsq);
  k_bnstats<<<(NC + 255)/256, 256, 0, stream>>>(gsum, gsq, lbn_w, lbn_b, scl, shf);
  k_bnapply<<<M2, 256, 0, stream>>>(fpre, scl, shf, fv, fvb);
  k_graph<<<B_, 256, 0, stream>>>(fv, Ag);
  k_gemm_vu<<<dim3(M2/128, 1024/128), 256, 0, stream>>>(fvb, wcat, bcat, vu);
  k_z<<<B_, 256, 0, stream>>>(vu, Ag, zg);
  k_bnv<<<N_, 256, 0, stream>>>(zg, bnv_w, bnv_b, bnv);
  k_normsc<<<N_ + NS_, 256, 0, stream>>>(main_sc, sub_sc, msc, ssc);
  k_final<<<M2, 256, 0, stream>>>(zg, fv, bnv, msc, ssc, out);
}
